// RPN_87299505258965
// MI455X (gfx1250) — matrix operand order repaired and checked by reading, not compiled, not run on hardware
//
#include <hip/hip_runtime.h>
#include <hip/hip_bf16.h>

typedef __bf16 bf16;
typedef __attribute__((ext_vector_type(16))) __bf16 v16bf;
typedef __attribute__((ext_vector_type(8)))  __bf16 v8bf;
typedef __attribute__((ext_vector_type(8)))  float  v8f;

#define HF 50
#define WF 84
#define HP 52
#define WP 86
#define CIN 512
#define KKC (CIN*9)        // 4608
#define KTC (KKC/32)       // 144
#define MM (HF*WF)         // 4200
#define MTILES ((MM+15)/16) // 263
#define NTILES (512/16)    // 32
#define AA 9
#define NANCH (MM*AA)      // 37800
#define PRE_NMS_N 1000
#define POST_NMS_N 300
#define SEL 2048
#define NBINS 65536
#define WIMF 1344.0f
#define HIMF 800.0f
#define LOGMAXF 4.135166556742356f
#define NMS_TH 0.7f

// ---------------- workspace layout ----------------
static constexpr size_t alup(size_t x){ return (x + 255) & ~(size_t)255; }
constexpr size_t OFF_XPAD = 0;                                             // bf16 [52][86][512] channel-last
constexpr size_t OFF_BP   = alup(OFF_XPAD + (size_t)CIN*HP*WP*2);          // bf16 conv W fragments
constexpr size_t OFF_WH   = alup(OFF_BP   + (size_t)KTC*NTILES*512*2);     // bf16 head W fragments
constexpr size_t OFF_CLS  = alup(OFF_WH   + (size_t)40960*2);              // f32 [4200][18]
constexpr size_t OFF_REG  = alup(OFF_CLS  + (size_t)MM*18*4);              // f32 [4200][36]
constexpr size_t OFF_PROP = alup(OFF_REG  + (size_t)MM*36*4);              // f32 [37800][4]
constexpr size_t OFF_SCOR = alup(OFF_PROP + (size_t)NANCH*4*4);            // f32 [37800]
constexpr size_t OFF_HIST = alup(OFF_SCOR + (size_t)NANCH*4);              // u32 [65536]
constexpr size_t OFF_META = alup(OFF_HIST + (size_t)NBINS*4);              // u32 [4]
constexpr size_t OFF_SELK = alup(OFF_META + 64);                           // u32 [SEL]
constexpr size_t OFF_SELI = alup(OFF_SELK + (size_t)SEL*4);                // u32 [SEL]
constexpr size_t OFF_TOPB = alup(OFF_SELI + (size_t)SEL*4);                // f32 [1000][4]
constexpr size_t OFF_TOPS = alup(OFF_TOPB + (size_t)PRE_NMS_N*4*4);        // f32 [1000]
constexpr size_t OFF_KEEP = alup(OFF_TOPS + (size_t)PRE_NMS_N*4);          // u32 [1000]

// ---------------- init / packing ----------------
__global__ void k_zero(unsigned* hist, unsigned* meta, unsigned* selK, unsigned* selI){
  int i = blockIdx.x*blockDim.x + threadIdx.x;
  if (i < NBINS) hist[i] = 0u;
  if (i < 4)     meta[i] = 0u;
  if (i < SEL){ selK[i] = 0u; selI[i] = 0u; }
}

// channel-last padded feature map: Xpad[(yy*WP+xx)*CIN + ic]
__global__ void k_pad(const float* __restrict__ fmap, bf16* __restrict__ Xpad){
  int idx = blockIdx.x*blockDim.x + threadIdx.x;
  if (idx >= HP*WP*CIN) return;
  int ic  = idx & (CIN-1);
  int pos = idx >> 9;
  int yy = pos / WP, xx = pos - yy*WP;
  float v = 0.0f;
  if (yy >= 1 && yy <= HF && xx >= 1 && xx <= WF)
    v = fmap[(size_t)ic*(HF*WF) + (yy-1)*WF + (xx-1)];
  Xpad[idx] = (bf16)v;
}

// conv W fragments, layout [kt][nt][lane][e].
// K permuted as K = tap*512 + ic  (tap = dy*3+dx); B uses contiguous-block layout.
__global__ void k_packconvw(const float* __restrict__ convw, bf16* __restrict__ Bp){
  int idx = blockIdx.x*blockDim.x + threadIdx.x;
  if (idx >= KTC*NTILES*512) return;
  int e = idx & 15, lane = (idx>>4) & 31, nt = (idx>>9) & 31, kt = idx >> 14;
  int N = nt*16 + (lane & 15);
  int K = kt*32 + e + ((lane >= 16) ? 16 : 0);
  int tap = K >> 9, ic = K & (CIN-1);
  int dy = tap/3, dx = tap - dy*3;
  Bp[idx] = (bf16)convw[(((size_t)N*CIN + ic)*3 + dy)*3 + dx];
}

// head weights: cls (16 ktiles x 2 ntiles) then reg (16 x 3), K = conv out channel (no permute)
__global__ void k_packhead(const float* __restrict__ clsw, const float* __restrict__ regw,
                           bf16* __restrict__ Wh){
  int idx = blockIdx.x*blockDim.x + threadIdx.x;
  if (idx >= 40960) return;
  if (idx < 16384){
    int e = idx & 15, lane = (idx>>4) & 31, nt = (idx>>9) & 1, kt = idx >> 10;
    int K  = kt*32 + e + ((lane >= 16) ? 16 : 0);
    int oc = nt*16 + (lane & 15);
    Wh[idx] = (bf16)((oc < 18) ? clsw[(size_t)oc*CIN + K] : 0.0f);
  } else {
    int j = idx - 16384;
    int e = j & 15, lane = (j>>4) & 31; int tt = j >> 9; int nt = tt % 3, kt = tt / 3;
    int K  = kt*32 + e + ((lane >= 16) ? 16 : 0);
    int oc = nt*16 + (lane & 15);
    Wh[idx] = (bf16)((oc < 36) ? regw[(size_t)oc*CIN + K] : 0.0f);
  }
}

// ---------------- fused conv3x3 + relu + 1x1 heads (WMMA) ----------------
__global__ void __launch_bounds__(128) k_conv(
    const float* __restrict__ convb, const float* __restrict__ clsb, const float* __restrict__ regb,
    const bf16* __restrict__ Xpad, const v16bf* __restrict__ Bp, const v16bf* __restrict__ Wh,
    float* __restrict__ clsOut, float* __restrict__ regOut)
{
  __shared__ __align__(32) bf16 lA[16*32*16];   // [kt2][lane][e] A-fragment staging, 16KB
  const int t = threadIdx.x;
  const int lane = t & 31, wv = t >> 5;
  const int mt = blockIdx.x;
  const int mloc = lane & 15;
  const int hi = lane >> 4;
  const int koff = hi * 16;                      // element offset of this lane-half
  int m = mt*16 + mloc; if (m >= MM) m = MM-1;
  const int y = m / WF, x = m - (m / WF) * WF;

  v8f acc[8] = {};
  for (int tap = 0; tap < 9; ++tap){
    const int dy = tap/3, dx = tap - (tap/3)*3;
    // A fragment for K-tile (tap, it): two contiguous 16B chunks per lane
    const bf16*  abase = Xpad + ((size_t)(y+dy)*WP + (x+dx))*CIN + koff;
    const v16bf* bbase = Bp + ((size_t)(tap*16)*NTILES + wv*8)*32 + lane;
    if (tap + 1 < 9){
      int dy2 = (tap+1)/3, dx2 = (tap+1) - ((tap+1)/3)*3;
      __builtin_prefetch((const void*)(Xpad + ((size_t)(y+dy2)*WP + (x+dx2))*CIN + koff), 0, 0);
    }
    for (int it = 0; it < 16; ++it){
      v8bf alo = *(const v8bf*)(abase + it*32);
      v8bf ahi = *(const v8bf*)(abase + it*32 + 8);
      v16bf a = __builtin_shufflevector(alo, ahi, 0,1,2,3,4,5,6,7,8,9,10,11,12,13,14,15);
      const v16bf* bput = bbase + (size_t)it*NTILES*32;
      #pragma unroll
      for (int n8 = 0; n8 < 8; ++n8){
        v16bf b = bput[(size_t)n8*32];
        acc[n8] = __builtin_amdgcn_wmma_f32_16x16x32_bf16(false, a, false, b,
                                                          (short)0, acc[n8], false, false);
      }
    }
  }

  // bias + relu, restage to LDS in A-fragment order (k of head GEMM = conv out channel)
  #pragma unroll
  for (int n8 = 0; n8 < 8; ++n8){
    int n = (wv*8 + n8)*16 + mloc;
    float bias = convb[n];
    int kt2 = n >> 5; int kkn = n & 31;
    int hb, e2;
    if      (kkn < 16){ hb=0; e2=kkn;    }
    else              { hb=1; e2=kkn-16; }
    #pragma unroll
    for (int r = 0; r < 8; ++r){
      int ml = r + hi*8;                   // D-tile row held by this lane
      float h = acc[n8][r] + bias;
      h = h > 0.0f ? h : 0.0f;
      int lane2 = ml + hb*16;
      lA[(kt2*32 + lane2)*16 + e2] = (bf16)h;
    }
  }
  __syncthreads();

  // head GEMM: tiles 0,1 = cls ntiles; 2,3,4 = reg ntiles
  const v16bf* lAv = (const v16bf*)lA;
  for (int tile = wv; tile < 5; tile += 4){
    v8f c2 = {};
    const v16bf* wbase; int ntH, isCls;
    if (tile < 2){ isCls = 1; ntH = tile;     wbase = Wh; }
    else         { isCls = 0; ntH = tile - 2; wbase = Wh + 16*2*32; }
    for (int k2 = 0; k2 < 16; ++k2){
      v16bf a2 = lAv[k2*32 + lane];
      v16bf b2 = wbase[((size_t)k2*(isCls ? 2 : 3) + ntH)*32 + lane];
      c2 = __builtin_amdgcn_wmma_f32_16x16x32_bf16(false, a2, false, b2,
                                                   (short)0, c2, false, false);
    }
    int col = ntH*16 + mloc;
    #pragma unroll
    for (int r = 0; r < 8; ++r){
      int ml = r + hi*8;
      int p  = mt*16 + ml;
      if (p < MM){
        if (isCls){ if (col < 18) clsOut[(size_t)p*18 + col] = c2[r] + clsb[col]; }
        else      { if (col < 36) regOut[(size_t)p*36 + col] = c2[r] + regb[col]; }
      }
    }
  }
}

// ---------------- proposals: anchors + regression + clip + score ----------------
__global__ void k_prop(const float* __restrict__ cls, const float* __restrict__ reg,
                       float* __restrict__ prop, float* __restrict__ scor){
  int i = blockIdx.x*blockDim.x + threadIdx.x;
  if (i >= NANCH) return;
  int p = i / AA, a = i - p*AA;
  int yy = p / WF, xx = p - yy*WF;
  const float scales[3] = {128.f, 256.f, 512.f};
  const float ratios[3] = {0.5f, 1.f, 2.f};
  int ri = a/3, si = a - ri*3;
  float rw  = sqrtf(ratios[ri]);
  float wsa = rw * scales[si];
  float hsa = scales[si] / rw;
  float sx = xx * 16.f, sy = yy * 16.f;
  float ax1 = sx + rintf(-wsa*0.5f), ay1 = sy + rintf(-hsa*0.5f);
  float ax2 = sx + rintf( wsa*0.5f), ay2 = sy + rintf( hsa*0.5f);
  float w = ax2-ax1, h = ay2-ay1;
  float cx = ax1 + 0.5f*w, cy = ay1 + 0.5f*h;
  const float* rg = reg + (size_t)p*36 + a*4;
  float dx = rg[0], dy = rg[1];
  float dw = fminf(rg[2], LOGMAXF), dh = fminf(rg[3], LOGMAXF);
  float pcx = dx*w + cx, pcy = dy*h + cy;
  float pw  = expf(dw)*w, ph  = expf(dh)*h;
  float x1 = fminf(fmaxf(pcx-0.5f*pw, 0.f), WIMF);
  float y1 = fminf(fmaxf(pcy-0.5f*ph, 0.f), HIMF);
  float x2 = fminf(fmaxf(pcx+0.5f*pw, 0.f), WIMF);
  float y2 = fminf(fmaxf(pcy+0.5f*ph, 0.f), HIMF);
  bool valid = (x2-x1 >= 16.f) && (y2-y1 >= 16.f);
  float l0 = cls[(size_t)p*18 + a*2], l1 = cls[(size_t)p*18 + a*2 + 1];
  float fg = 1.f/(1.f + expf(l0 - l1));
  float s = valid ? fg : -1.f;
  prop[(size_t)i*4+0]=x1; prop[(size_t)i*4+1]=y1;
  prop[(size_t)i*4+2]=x2; prop[(size_t)i*4+3]=y2;
  scor[i] = s;
}

// ---------------- top-1000 radix select + bitonic sort ----------------
__device__ __forceinline__ unsigned fkey(float s){
  unsigned u = __float_as_uint(s);
  return (u & 0x80000000u) ? ~u : (u | 0x80000000u);
}
__global__ void k_hist(const float* __restrict__ scor, unsigned* __restrict__ hist){
  int i = blockIdx.x*blockDim.x + threadIdx.x;
  if (i >= NANCH) return;
  atomicAdd(&hist[fkey(scor[i]) >> 16], 1u);
}
__global__ void k_scan(const unsigned* __restrict__ hist, unsigned* __restrict__ meta){
  if (threadIdx.x == 0 && blockIdx.x == 0){
    unsigned cum = 0; int b = NBINS-1;
    for (; b >= 0; --b){ unsigned c = hist[b]; if (cum + c >= PRE_NMS_N) break; cum += c; }
    if (b < 0) b = 0;
    meta[0] = (unsigned)b; meta[1] = cum;  // C1 = count strictly above bin
  }
}
__global__ void k_compact(const float* __restrict__ scor, unsigned* __restrict__ meta,
                          unsigned* __restrict__ selK, unsigned* __restrict__ selI){
  int i = blockIdx.x*blockDim.x + threadIdx.x;
  if (i >= NANCH) return;
  unsigned key = fkey(scor[i]); unsigned bin = key >> 16;
  unsigned B = meta[0], C1 = meta[1];
  if (bin > B){
    unsigned pos = atomicAdd(&meta[2], 1u);
    if (pos < SEL){ selK[pos] = key; selI[pos] = i; }
  } else if (bin == B){
    unsigned pos = C1 + atomicAdd(&meta[3], 1u);
    if (pos < SEL){ selK[pos] = key; selI[pos] = i; }
  }
}
__global__ void __launch_bounds__(1024) k_sortsel(
    const unsigned* __restrict__ selK, const unsigned* __restrict__ selI,
    const float* __restrict__ prop, const float* __restrict__ scor,
    float* __restrict__ topB, float* __restrict__ topS){
  __shared__ unsigned long long kk[SEL];
  __shared__ unsigned ii[SEL];
  int t = threadIdx.x;
  for (int s = t; s < SEL; s += 1024){
    unsigned gi = selI[s];
    kk[s] = ((unsigned long long)selK[s] << 32) | (unsigned long long)(0xFFFFFFFFu - gi);
    ii[s] = gi;
  }
  __syncthreads();
  for (int k = 2; k <= SEL; k <<= 1){
    for (int j = k >> 1; j > 0; j >>= 1){
      for (int s = t; s < SEL; s += 1024){
        int xp = s ^ j;
        if (xp > s){
          bool desc = ((s & k) == 0);
          unsigned long long A = kk[s], Bv = kk[xp];
          if ((A < Bv) == desc){
            kk[s] = Bv; kk[xp] = A;
            unsigned tmp = ii[s]; ii[s] = ii[xp]; ii[xp] = tmp;
          }
        }
      }
      __syncthreads();
    }
  }
  if (t < PRE_NMS_N){
    unsigned g = ii[t];
    topB[t*4+0] = prop[(size_t)g*4+0]; topB[t*4+1] = prop[(size_t)g*4+1];
    topB[t*4+2] = prop[(size_t)g*4+2]; topB[t*4+3] = prop[(size_t)g*4+3];
    topS[t] = scor[g];
  }
}

// ---------------- NMS (serial-i, parallel-j, LDS-resident) ----------------
__global__ void __launch_bounds__(1024) k_nms(const float* __restrict__ topB,
                                              unsigned* __restrict__ keepG){
  __shared__ float bx1[PRE_NMS_N], by1[PRE_NMS_N], bx2[PRE_NMS_N], by2[PRE_NMS_N];
  __shared__ float area[PRE_NMS_N];
  __shared__ unsigned kp[PRE_NMS_N];
  int t = threadIdx.x;
  for (int i = t; i < PRE_NMS_N; i += 1024){
    float x1 = topB[i*4+0], y1 = topB[i*4+1], x2 = topB[i*4+2], y2 = topB[i*4+3];
    bx1[i]=x1; by1[i]=y1; bx2[i]=x2; by2[i]=y2;
    area[i] = (x2-x1)*(y2-y1);
    kp[i] = 1u;
  }
  __syncthreads();
  for (int i = 0; i < PRE_NMS_N-1; ++i){
    if (t > i && t < PRE_NMS_N && kp[i]){
      float xx1 = fmaxf(bx1[i], bx1[t]);
      float yy1 = fmaxf(by1[i], by1[t]);
      float xx2 = fminf(bx2[i], bx2[t]);
      float yy2 = fminf(by2[i], by2[t]);
      float inter = fmaxf(xx2-xx1, 0.f) * fmaxf(yy2-yy1, 0.f);
      float iou = inter / (area[i] + area[t] - inter);
      if (iou > NMS_TH) kp[t] = 0u;
    }
    __syncthreads();
  }
  for (int i = t; i < PRE_NMS_N; i += 1024) keepG[i] = kp[i];
}

// ---------------- final top-300 of where(keep, s, -1) ----------------
__global__ void k_final(const unsigned* __restrict__ keep, const float* __restrict__ topB,
                        const float* __restrict__ topS, float* __restrict__ out){
  if (threadIdx.x != 0 || blockIdx.x != 0) return;
  int cnt = 0;
  for (int pass = 0; pass < 2 && cnt < POST_NMS_N; ++pass){
    for (int i = 0; i < PRE_NMS_N && cnt < POST_NMS_N; ++i){
      bool k = keep[i] != 0u;
      if (pass == 0 ? k : !k){
        out[cnt*4+0] = topB[i*4+0]; out[cnt*4+1] = topB[i*4+1];
        out[cnt*4+2] = topB[i*4+2]; out[cnt*4+3] = topB[i*4+3];
        out[4*POST_NMS_N + cnt] = topS[i];
        ++cnt;
      }
    }
  }
}

// ---------------- host launch ----------------
extern "C" void kernel_launch(void* const* d_in, const int* in_sizes, int n_in,
                              void* d_out, int out_size, void* d_ws, size_t ws_size,
                              hipStream_t stream){
  (void)in_sizes; (void)n_in; (void)out_size; (void)ws_size;
  const float* fmap  = (const float*)d_in[1];
  const float* convw = (const float*)d_in[2];
  const float* convb = (const float*)d_in[3];
  const float* clsw  = (const float*)d_in[4];
  const float* clsb  = (const float*)d_in[5];
  const float* regw  = (const float*)d_in[6];
  const float* regb  = (const float*)d_in[7];
  char* ws = (char*)d_ws;

  bf16*     Xpad = (bf16*)    (ws + OFF_XPAD);
  bf16*     Bp   = (bf16*)    (ws + OFF_BP);
  bf16*     Wh   = (bf16*)    (ws + OFF_WH);
  float*    clsO = (float*)   (ws + OFF_CLS);
  float*    regO = (float*)   (ws + OFF_REG);
  float*    prop = (float*)   (ws + OFF_PROP);
  float*    scor = (float*)   (ws + OFF_SCOR);
  unsigned* hist = (unsigned*)(ws + OFF_HIST);
  unsigned* meta = (unsigned*)(ws + OFF_META);
  unsigned* selK = (unsigned*)(ws + OFF_SELK);
  unsigned* selI = (unsigned*)(ws + OFF_SELI);
  float*    topB = (float*)   (ws + OFF_TOPB);
  float*    topS = (float*)   (ws + OFF_TOPS);
  unsigned* keep = (unsigned*)(ws + OFF_KEEP);
  float*    out  = (float*)d_out;

  k_zero    <<<(NBINS+255)/256, 256, 0, stream>>>(hist, meta, selK, selI);
  k_pad     <<<(HP*WP*CIN+255)/256, 256, 0, stream>>>(fmap, Xpad);
  k_packconvw<<<(KTC*NTILES*512+255)/256, 256, 0, stream>>>(convw, Bp);
  k_packhead<<<(40960+255)/256, 256, 0, stream>>>(clsw, regw, Wh);
  k_conv    <<<MTILES, 128, 0, stream>>>(convb, clsb, regb, Xpad,
                                         (const v16bf*)Bp, (const v16bf*)Wh, clsO, regO);
  k_prop    <<<(NANCH+255)/256, 256, 0, stream>>>(clsO, regO, prop, scor);
  k_hist    <<<(NANCH+255)/256, 256, 0, stream>>>(scor, hist);
  k_scan    <<<1, 32, 0, stream>>>(hist, meta);
  k_compact <<<(NANCH+255)/256, 256, 0, stream>>>(scor, meta, selK, selI);
  k_sortsel <<<1, 1024, 0, stream>>>(selK, selI, prop, scor, topB, topS);
  k_nms     <<<1, 1024, 0, stream>>>(topB, keep);
  k_final   <<<1, 32, 0, stream>>>(keep, topB, topS, out);
}
